// MoELayer_15040975470727
// MI455X (gfx1250) — matrix operand order repaired and checked by reading, not compiled, not run on hardware
//
#include <hip/hip_runtime.h>

#define N_TOKENS  8192
#define D_IN      2048
#define D_OUT     2048
#define N_EXPERTS 8

typedef __attribute__((ext_vector_type(16))) __bf16       v16bf;
typedef __attribute__((ext_vector_type(8)))  float        v8f;
typedef __attribute__((ext_vector_type(4)))  unsigned int u32x4;

union FragBF {
    v16bf v;
    u32x4 q[2];
};

__device__ __forceinline__ unsigned short f2bf(float f) {
    unsigned int u = __builtin_bit_cast(unsigned int, f);
    u += 0x7FFFu + ((u >> 16) & 1u);            // round-to-nearest-even
    return (unsigned short)(u >> 16);
}
__device__ __forceinline__ unsigned int pack2bf(float lo, float hi) {
    return (unsigned int)f2bf(lo) | ((unsigned int)f2bf(hi) << 16);
}

// ---------------------------------------------------------------- init
__global__ void moe_init(int* __restrict__ counts) {
    if (threadIdx.x < N_EXPERTS) counts[threadIdx.x] = 0;
}

// ---------------------------------------------------------------- gate
// One wave32 per token: logits over 8 experts, softmax, argmax,
// per-expert atomic position assignment.
__global__ __launch_bounds__(256)
void moe_gate(const float* __restrict__ x,  const float* __restrict__ Wg,
              const float* __restrict__ bg,
              float* __restrict__ gate_probs, float* __restrict__ selected_out,
              int* __restrict__ sel_idx, int* __restrict__ pos_in_expert,
              int* __restrict__ counts) {
    const int lane = threadIdx.x & 31;
    const int wid  = threadIdx.x >> 5;
    const int t    = blockIdx.x * 8 + wid;

    const float4* xr = (const float4*)(x + (size_t)t * D_IN);
    float acc[N_EXPERTS];
#pragma unroll
    for (int e = 0; e < N_EXPERTS; ++e) acc[e] = 0.0f;

    for (int i = 0; i < D_IN / 128; ++i) {
        float4 xv = xr[lane + 32 * i];
#pragma unroll
        for (int e = 0; e < N_EXPERTS; ++e) {
            float4 wv = ((const float4*)(Wg + (size_t)e * D_IN))[lane + 32 * i];
            acc[e] += xv.x * wv.x + xv.y * wv.y + xv.z * wv.z + xv.w * wv.w;
        }
    }
#pragma unroll
    for (int e = 0; e < N_EXPERTS; ++e)
        for (int off = 16; off > 0; off >>= 1)
            acc[e] += __shfl_xor(acc[e], off, 32);

    // every lane now holds the full reduced logits; lane 0 finalizes
    if (lane == 0) {
        float lg[N_EXPERTS];
        float mx = -3.4e38f;
#pragma unroll
        for (int e = 0; e < N_EXPERTS; ++e) { lg[e] = acc[e] + bg[e]; mx = fmaxf(mx, lg[e]); }
        float p[N_EXPERTS];
        float sum = 0.0f;
#pragma unroll
        for (int e = 0; e < N_EXPERTS; ++e) { p[e] = __expf(lg[e] - mx); sum += p[e]; }
        const float inv = 1.0f / sum;
        int   best = 0;
        float bv   = p[0];
#pragma unroll
        for (int e = 0; e < N_EXPERTS; ++e) {
            p[e] *= inv;
            gate_probs[(size_t)t * N_EXPERTS + e] = p[e];
            if (p[e] > bv) { bv = p[e]; best = e; }   // strict > keeps first max
        }
        selected_out[t]   = (float)best;
        sel_idx[t]        = best;
        pos_in_expert[t]  = atomicAdd(&counts[best], 1);
    }
}

// ---------------------------------------------------------------- scan
__global__ void moe_scan(const int* __restrict__ counts, int* __restrict__ offsets,
                         float* __restrict__ usage_out) {
    if (threadIdx.x == 0 && blockIdx.x == 0) {
        int off = 0;
        for (int e = 0; e < N_EXPERTS; ++e) {
            offsets[e]   = off;
            off         += counts[e];
            usage_out[e] = (float)counts[e];
        }
    }
}

// ---------------------------------------------------------------- scatter
__global__ void moe_scatter(const int* __restrict__ sel_idx, const int* __restrict__ pos,
                            const int* __restrict__ offsets, int* __restrict__ token_list) {
    int t = blockIdx.x * 256 + threadIdx.x;
    if (t < N_TOKENS) token_list[offsets[sel_idx[t]] + pos[t]] = t;
}

// ---------------------------------------------------------------- f32 -> bf16
__global__ void moe_cvt_bf16(const float* __restrict__ src, unsigned int* __restrict__ dst,
                             int n8) {
    int i = blockIdx.x * 256 + threadIdx.x;
    if (i >= n8) return;
    const float4* p = (const float4*)src + (size_t)i * 2;
    float4 a = p[0], b = p[1];
    u32x4 o;
    o[0] = pack2bf(a.x, a.y);
    o[1] = pack2bf(a.z, a.w);
    o[2] = pack2bf(b.x, b.y);
    o[3] = pack2bf(b.z, b.w);
    ((u32x4*)dst)[i] = o;
}

// ---------------------------------------------------------------- grouped GEMM
// block = 256 threads (8 waves) arranged 4(M) x 2(N) over a 128x128 block tile.
// Per-wave tile = 32 gathered tokens x 64 out cols; waves sharing an A row-group
// or B col-group dedupe their fragment loads in the WGP$ (3x less L2 traffic).
// D = A(Xg 16x32 bf16) x B(We^T 32x16 bf16) + C, K looped over 2048.
__global__ __launch_bounds__(256)
void moe_gemm(const unsigned short* __restrict__ x_bf,
              const unsigned short* __restrict__ we_bf,
              const float* __restrict__ be,
              const int* __restrict__ token_list,
              const int* __restrict__ counts,
              const int* __restrict__ offsets,
              float* __restrict__ out) {
    const int e = blockIdx.z;
    const int cnt = counts[e];
    if ((int)blockIdx.y * 128 >= cnt) return;  // uniform exit: EXEC stays all-ones
    const int off_e = offsets[e];

    const int lane = threadIdx.x & 31;
    const int wave = threadIdx.x >> 5;         // 0..7
    const int wm   = wave >> 1;                // 0..3 : M sub-tile
    const int wn   = wave & 1;                 // 0..1 : N sub-tile
    const int lo   = lane & 15;
    const int hi   = lane >> 4;
    const int m_base   = blockIdx.y * 128 + wm * 32;
    const int col_base = blockIdx.x * 128 + wn * 64;

    // A rows: gathered tokens for the two 16-row groups (clamped; masked at store)
    const int m0 = m_base + lo;
    const int m1 = m_base + 16 + lo;
    const int tok0 = token_list[off_e + (m0 < cnt ? m0 : cnt - 1)];
    const int tok1 = token_list[off_e + (m1 < cnt ? m1 : cnt - 1)];
    // A layout (16-bit 16x32): half h covers K = {8h..8h+7, 16+8h..16+8h+7}
    const unsigned short* arow0 = x_bf + (size_t)tok0 * D_IN + 8 * hi;
    const unsigned short* arow1 = x_bf + (size_t)tok1 * D_IN + 8 * hi;
    // B layout (16-bit 32x16): half h covers K = {8h..8h+7, 16+8h..16+8h+7}
    const unsigned short* brow[4];
#pragma unroll
    for (int ng = 0; ng < 4; ++ng)
        brow[ng] = we_bf + ((size_t)e * D_OUT + col_base + ng * 16 + lo) * D_IN + 8 * hi;

    v8f acc[8];
#pragma unroll
    for (int i = 0; i < 8; ++i) acc[i] = (v8f){0, 0, 0, 0, 0, 0, 0, 0};

    for (int kb = 0; kb < D_IN; kb += 32) {
        FragBF a0, a1, b[4];
        a0.q[0] = *(const u32x4*)(arow0 + kb);
        a0.q[1] = *(const u32x4*)(arow0 + kb + 16);
        a1.q[0] = *(const u32x4*)(arow1 + kb);
        a1.q[1] = *(const u32x4*)(arow1 + kb + 16);
#pragma unroll
        for (int ng = 0; ng < 4; ++ng) {
            b[ng].q[0] = *(const u32x4*)(brow[ng] + kb);
            b[ng].q[1] = *(const u32x4*)(brow[ng] + kb + 16);
        }
#pragma unroll
        for (int ng = 0; ng < 4; ++ng) {
            acc[ng]     = __builtin_amdgcn_wmma_f32_16x16x32_bf16(
                              false, a0.v, false, b[ng].v, (short)0, acc[ng], false, false);
            acc[4 + ng] = __builtin_amdgcn_wmma_f32_16x16x32_bf16(
                              false, a1.v, false, b[ng].v, (short)0, acc[4 + ng], false, false);
        }
    }

    float bev[4];
#pragma unroll
    for (int ng = 0; ng < 4; ++ng)
        bev[ng] = be[(size_t)e * D_OUT + col_base + ng * 16 + lo];

#pragma unroll
    for (int mg = 0; mg < 2; ++mg) {
#pragma unroll
        for (int d = 0; d < 8; ++d) {
            const int m = m_base + mg * 16 + 8 * hi + d;   // C/D: elem d -> row d+8*hi
            if (m < cnt) {
                const int tok = token_list[off_e + m];
                float* orow = out + (size_t)tok * D_OUT + col_base + lo;
#pragma unroll
                for (int ng = 0; ng < 4; ++ng)
                    orow[ng * 16] = acc[mg * 4 + ng][d] + bev[ng];
            }
        }
    }
}

// ---------------------------------------------------------------- launch
extern "C" void kernel_launch(void* const* d_in, const int* in_sizes, int n_in,
                              void* d_out, int out_size, void* d_ws, size_t ws_size,
                              hipStream_t stream) {
    const float* x  = (const float*)d_in[0];
    const float* Wg = (const float*)d_in[1];
    const float* bg = (const float*)d_in[2];
    const float* We = (const float*)d_in[3];
    const float* be = (const float*)d_in[4];

    float* out        = (float*)d_out;                               // [8192,2048]
    float* gate_probs = out + (size_t)N_TOKENS * D_OUT;              // [8192,8]
    float* selected   = gate_probs + (size_t)N_TOKENS * N_EXPERTS;   // [8192,1] (as float)
    float* usage      = selected + N_TOKENS;                         // [8]

    // workspace: x_bf16 (32MB) | We_bf16 (64MB) | ints (~100KB)
    char* ws = (char*)d_ws;
    unsigned short* x_bf  = (unsigned short*)ws;
    unsigned short* we_bf = (unsigned short*)(ws + (size_t)N_TOKENS * D_IN * 2);
    int* ip         = (int*)(ws + (size_t)N_TOKENS * D_IN * 2
                                + (size_t)N_EXPERTS * D_OUT * D_IN * 2);
    int* counts     = ip;                       // 8
    int* offsets    = counts + N_EXPERTS;       // 8
    int* sel_idx    = offsets + N_EXPERTS;      // 8192
    int* pos        = sel_idx + N_TOKENS;       // 8192
    int* token_list = pos + N_TOKENS;           // 8192

    moe_init<<<1, 32, 0, stream>>>(counts);
    moe_gate<<<N_TOKENS / 8, 256, 0, stream>>>(x, Wg, bg, gate_probs, selected,
                                               sel_idx, pos, counts);
    moe_scan<<<1, 32, 0, stream>>>(counts, offsets, usage);
    moe_scatter<<<N_TOKENS / 256, 256, 0, stream>>>(sel_idx, pos, offsets, token_list);

    const int nx8 = N_TOKENS * D_IN / 8;
    moe_cvt_bf16<<<(nx8 + 255) / 256, 256, 0, stream>>>(x, (unsigned int*)x_bf, nx8);
    const int nw8 = N_EXPERTS * D_OUT * D_IN / 8;
    moe_cvt_bf16<<<(nw8 + 255) / 256, 256, 0, stream>>>(We, (unsigned int*)we_bf, nw8);

    dim3 grid(D_OUT / 128, N_TOKENS / 128, N_EXPERTS);
    moe_gemm<<<grid, 256, 0, stream>>>(x_bf, we_bf, be, token_list, counts, offsets, out);
}
